// EncoderGNN_50663434224157
// MI455X (gfx1250) — matrix operand order repaired and checked by reading, not compiled, not run on hardware
//
#include <hip/hip_runtime.h>
#include <hip/hip_bf16.h>

#define DFEAT   256
#define KCAT    512
#define NSPATIAL 65160
#define NSPHERE  40962
#define NTOT    (NSPATIAL + NSPHERE)   // 106122
#define NEDGE   400000
#define NSTEP   3
#define MROWS   32                      // rows (edges/nodes) per block

typedef __attribute__((ext_vector_type(16))) __bf16 v16bf;
typedef __attribute__((ext_vector_type(8)))  float  v8f;

struct alignas(16) U4 { unsigned int x, y, z, w; };
union BFrag { v16bf v; U4 u[2]; };

__device__ __forceinline__ unsigned short f2bf(float f) {
    unsigned int u = __float_as_uint(f);
    u += 0x7fffu + ((u >> 16) & 1u);   // round-to-nearest-even
    return (unsigned short)(u >> 16);
}

// ---------------------------------------------------------------------------
// Convert fp32 weights -> bf16, transposed to [out_ch][K] row-major.
// which: 0=W1e(K=512, src row 3+k), 1=W2e(K=256), 2=W1n(K=512), 3=W2n(K=256)
// grid = dim3(512, 3, 4), block = 256
// ---------------------------------------------------------------------------
__global__ void convert_weights(const float* __restrict__ W1e,
                                const float* __restrict__ W2e,
                                const float* __restrict__ W1n,
                                const float* __restrict__ W2n,
                                unsigned short* __restrict__ w1e_t,
                                unsigned short* __restrict__ w2e_t,
                                unsigned short* __restrict__ w1n_t,
                                unsigned short* __restrict__ w2n_t) {
    const int s     = blockIdx.y;
    const int which = blockIdx.z;
    const int idx   = blockIdx.x * 256 + threadIdx.x;
    if (which == 0) {                      // [256][512]: w[n][k] = W1e[s][3+k][n]
        int n = idx >> 9, k = idx & 511;
        w1e_t[(size_t)s * 131072 + idx] =
            f2bf(W1e[(size_t)s * 515 * 256 + (size_t)(3 + k) * 256 + n]);
    } else if (which == 1) {               // [256][256]: w[n][k] = W2e[s][k][n]
        if (idx < 65536) {
            int n = idx >> 8, k = idx & 255;
            w2e_t[(size_t)s * 65536 + idx] =
                f2bf(W2e[(size_t)s * 65536 + (size_t)k * 256 + n]);
        }
    } else if (which == 2) {               // [256][512]: w[n][k] = W1n[s][k][n]
        int n = idx >> 9, k = idx & 511;
        w1n_t[(size_t)s * 131072 + idx] =
            f2bf(W1n[(size_t)s * 131072 + (size_t)k * 256 + n]);
    } else {                               // [256][256]
        if (idx < 65536) {
            int n = idx >> 8, k = idx & 255;
            w2n_t[(size_t)s * 65536 + idx] =
                f2bf(W2n[(size_t)s * 65536 + (size_t)k * 256 + n]);
        }
    }
}

__global__ void zero_f4(float4* __restrict__ p, int n4) {
    int i = blockIdx.x * 256 + threadIdx.x;
    if (i < n4) p[i] = make_float4(0.f, 0.f, 0.f, 0.f);
}

// ---------------------------------------------------------------------------
// Fragment loaders. A-matrix (16x32 bf16) per ISA layout:
//   lane L(0-15): row = L, K chunks {0..7} and {16..23}
//   lane L(16-31): row = L-16, K chunks {8..15} and {24..31}
// B-matrix (32x16 bf16): lane L: col = L&15, K chunks {(L>>4)*8 + 0..7} and {16 + (L>>4)*8 + 0..7}.
// ---------------------------------------------------------------------------
__device__ __forceinline__ BFrag load_a_frag(const unsigned short* p16) {
    BFrag a;
    a.u[0] = *(const U4*)(p16);
    a.u[1] = *(const U4*)(p16 + 16);
    return a;
}
__device__ __forceinline__ BFrag load_b_frag(const unsigned short* p16) {
    BFrag b;
    b.u[0] = *(const U4*)(p16);
    b.u[1] = *(const U4*)(p16 + 16);
    return b;
}

// ---------------------------------------------------------------------------
// Edge MLP + scatter. 32 edges per block, 128 threads (4 waves).
// Wave w owns output columns [w*64, w*64+64) for all 32 rows:
// 2 (M-tiles) x 4 (N-tiles) accumulators; each B fragment feeds 2 WMMAs.
// ---------------------------------------------------------------------------
__launch_bounds__(128)
__global__ void edge_mlp_scatter(const float* __restrict__ nodes,
                                 const float* __restrict__ edges,
                                 const int*   __restrict__ senders,
                                 const int*   __restrict__ receivers,
                                 const unsigned short* __restrict__ W1t,   // [256][512]
                                 const float* __restrict__ W1edge,         // rows 0..2 of W1e
                                 const float* __restrict__ b1,
                                 const float* __restrict__ gamma,
                                 const float* __restrict__ beta,
                                 const unsigned short* __restrict__ W2t,   // [256][256]
                                 const float* __restrict__ b2,
                                 float* __restrict__ agg) {
    // sA (bf16 [32][520]) and sH (f32 [32][260]) alias the same 33.3 KB region:
    // sA is dead after GEMM1 (separated by __syncthreads()).
    __shared__ __align__(16) char smemAH[MROWS * 1040];
    unsigned short (*sA)[KCAT + 8] = (unsigned short (*)[KCAT + 8])smemAH;
    float          (*sH)[DFEAT + 4] = (float (*)[DFEAT + 4])smemAH;
    __shared__ unsigned short sHb[MROWS][DFEAT + 8];
    __shared__ float          sE[MROWS][3];
    __shared__ int            sSend[MROWS];
    __shared__ int            sRecv[MROWS];
    __shared__ float          sRed[MROWS][4][2];
    __shared__ float          sMu[MROWS];
    __shared__ float          sRs[MROWS];

    const int tid = threadIdx.x;
    const int e0  = blockIdx.x * MROWS;

    if (tid < MROWS) {
        int e = e0 + tid;
        sSend[tid] = senders[e];
        sRecv[tid] = receivers[e];
        sE[tid][0] = edges[e * 3 + 0];
        sE[tid][1] = edges[e * 3 + 1];
        sE[tid][2] = edges[e * 3 + 2];
    }
    __syncthreads();

    // Stage gathered [sender | receiver] rows as bf16: 32 rows x 512 cols.
    for (int i = tid; i < MROWS * 128; i += 128) {
        int row  = i >> 7;
        int q    = i & 127;
        int part = q >> 6;          // 0 = sender half, 1 = receiver half
        int f4   = q & 63;
        int nd   = part ? sRecv[row] : sSend[row];
        const float4 val = *(const float4*)(nodes + (size_t)nd * DFEAT + f4 * 4);
        unsigned short* dst = &sA[row][part * 256 + f4 * 4];
        dst[0] = f2bf(val.x); dst[1] = f2bf(val.y);
        dst[2] = f2bf(val.z); dst[3] = f2bf(val.w);
    }
    __syncthreads();

    const int lane = tid & 31;
    const int wave = tid >> 5;
    const int r16  = lane & 15;
    const int hi   = lane >> 4;

    v8f acc[2][4];
#pragma unroll
    for (int m = 0; m < 2; ++m)
#pragma unroll
        for (int t = 0; t < 4; ++t)
            acc[m][t] = v8f{0.f,0.f,0.f,0.f,0.f,0.f,0.f,0.f};

    // GEMM1: [32x512] x [512x256] (bf16 -> f32); B frag reused for 2 M-tiles.
    for (int kb = 0; kb < KCAT; kb += 32) {
        BFrag a0 = load_a_frag(&sA[r16][kb + hi * 8]);
        BFrag a1 = load_a_frag(&sA[16 + r16][kb + hi * 8]);
#pragma unroll
        for (int t = 0; t < 4; ++t) {
            const int n = wave * 64 + t * 16 + r16;
            BFrag b = load_b_frag(W1t + (size_t)n * KCAT + kb + hi * 8);
            acc[0][t] = __builtin_amdgcn_wmma_f32_16x16x32_bf16(
                false, a0.v, false, b.v, (short)0, acc[0][t], false, false);
            acc[1][t] = __builtin_amdgcn_wmma_f32_16x16x32_bf16(
                false, a1.v, false, b.v, (short)0, acc[1][t], false, false);
        }
    }
    __syncthreads();   // sA dead; sH (aliased) can be written now

    // Epilogue: raw-edge(3) columns + bias + relu -> sH
#pragma unroll
    for (int t = 0; t < 4; ++t) {
        const int c  = wave * 64 + t * 16 + r16;
        const float w0 = W1edge[0 * DFEAT + c];
        const float w1 = W1edge[1 * DFEAT + c];
        const float w2 = W1edge[2 * DFEAT + c];
        const float bb = b1[c];
#pragma unroll
        for (int m = 0; m < 2; ++m) {
#pragma unroll
            for (int v = 0; v < 8; ++v) {
                const int r = m * 16 + v + hi * 8;
                float x = acc[m][t][v] + sE[r][0] * w0 + sE[r][1] * w1 + sE[r][2] * w2 + bb;
                sH[r][c] = x > 0.f ? x : 0.f;
            }
        }
    }
    __syncthreads();

    // LayerNorm over D=256 per row (4 threads per row, 64 cols each).
    {
        const int r = tid >> 2, sub = tid & 3;
        float sum = 0.f, sq = 0.f;
        for (int c = sub * 64; c < sub * 64 + 64; ++c) {
            float x = sH[r][c]; sum += x; sq += x * x;
        }
        sRed[r][sub][0] = sum; sRed[r][sub][1] = sq;
    }
    __syncthreads();
    if (tid < MROWS) {
        float sum = 0.f, sq = 0.f;
        for (int i = 0; i < 4; ++i) { sum += sRed[tid][i][0]; sq += sRed[tid][i][1]; }
        float mu  = sum * (1.f / DFEAT);
        float var = sq * (1.f / DFEAT) - mu * mu;
        sMu[tid] = mu;
        sRs[tid] = rsqrtf(var + 1e-5f);
    }
    __syncthreads();
    for (int i = tid; i < MROWS * DFEAT; i += 128) {
        int r = i >> 8, c = i & 255;
        float x = (sH[r][c] - sMu[r]) * sRs[r] * gamma[c] + beta[c];
        sHb[r][c] = f2bf(x);
    }
    __syncthreads();

    // GEMM2: [32x256] x [256x256]
    v8f acc2[2][4];
#pragma unroll
    for (int m = 0; m < 2; ++m)
#pragma unroll
        for (int t = 0; t < 4; ++t)
            acc2[m][t] = v8f{0.f,0.f,0.f,0.f,0.f,0.f,0.f,0.f};
    for (int kb = 0; kb < DFEAT; kb += 32) {
        BFrag a0 = load_a_frag(&sHb[r16][kb + hi * 8]);
        BFrag a1 = load_a_frag(&sHb[16 + r16][kb + hi * 8]);
#pragma unroll
        for (int t = 0; t < 4; ++t) {
            const int n = wave * 64 + t * 16 + r16;
            BFrag b = load_b_frag(W2t + (size_t)n * DFEAT + kb + hi * 8);
            acc2[0][t] = __builtin_amdgcn_wmma_f32_16x16x32_bf16(
                false, a0.v, false, b.v, (short)0, acc2[0][t], false, false);
            acc2[1][t] = __builtin_amdgcn_wmma_f32_16x16x32_bf16(
                false, a1.v, false, b.v, (short)0, acc2[1][t], false, false);
        }
    }

    // Scatter-add messages into agg[receiver] (L2-resident fp32 atomics).
#pragma unroll
    for (int t = 0; t < 4; ++t) {
        const int c  = wave * 64 + t * 16 + r16;
        const float bb = b2[c];
#pragma unroll
        for (int m = 0; m < 2; ++m) {
#pragma unroll
            for (int v = 0; v < 8; ++v) {
                const int r = m * 16 + v + hi * 8;
                atomicAdd(&agg[(size_t)sRecv[r] * DFEAT + c], acc2[m][t][v] + bb);
            }
        }
    }
}

// ---------------------------------------------------------------------------
// Node MLP: 32 nodes per block, K = [nodes | agg] = 512. In-place safe
// (row n output depends only on row n inputs, read before write per block).
// ---------------------------------------------------------------------------
__launch_bounds__(128)
__global__ void node_mlp(const float* __restrict__ nodes_in,
                         const float* __restrict__ agg,
                         const unsigned short* __restrict__ W1t,   // [256][512]
                         const float* __restrict__ b1,
                         const float* __restrict__ gamma,
                         const float* __restrict__ beta,
                         const unsigned short* __restrict__ W2t,   // [256][256]
                         const float* __restrict__ b2,
                         float* __restrict__ nodes_out) {
    __shared__ __align__(16) char smemAH[MROWS * 1040];
    unsigned short (*sA)[KCAT + 8] = (unsigned short (*)[KCAT + 8])smemAH;
    float          (*sH)[DFEAT + 4] = (float (*)[DFEAT + 4])smemAH;
    __shared__ unsigned short sHb[MROWS][DFEAT + 8];
    __shared__ float          sRed[MROWS][4][2];
    __shared__ float          sMu[MROWS];
    __shared__ float          sRs[MROWS];

    const int tid = threadIdx.x;
    const int n0  = blockIdx.x * MROWS;

    for (int i = tid; i < MROWS * 128; i += 128) {
        int row  = i >> 7;
        int q    = i & 127;
        int part = q >> 6;          // 0 = node feats, 1 = agg
        int f4   = q & 63;
        int nd   = n0 + row;
        float4 val = make_float4(0.f, 0.f, 0.f, 0.f);
        if (nd < NTOT) {
            const float* src = part ? (agg + (size_t)nd * DFEAT)
                                    : (nodes_in + (size_t)nd * DFEAT);
            val = *(const float4*)(src + f4 * 4);
        }
        unsigned short* dst = &sA[row][part * 256 + f4 * 4];
        dst[0] = f2bf(val.x); dst[1] = f2bf(val.y);
        dst[2] = f2bf(val.z); dst[3] = f2bf(val.w);
    }
    __syncthreads();

    const int lane = tid & 31;
    const int wave = tid >> 5;
    const int r16  = lane & 15;
    const int hi   = lane >> 4;

    v8f acc[2][4];
#pragma unroll
    for (int m = 0; m < 2; ++m)
#pragma unroll
        for (int t = 0; t < 4; ++t)
            acc[m][t] = v8f{0.f,0.f,0.f,0.f,0.f,0.f,0.f,0.f};
    for (int kb = 0; kb < KCAT; kb += 32) {
        BFrag a0 = load_a_frag(&sA[r16][kb + hi * 8]);
        BFrag a1 = load_a_frag(&sA[16 + r16][kb + hi * 8]);
#pragma unroll
        for (int t = 0; t < 4; ++t) {
            const int n = wave * 64 + t * 16 + r16;
            BFrag b = load_b_frag(W1t + (size_t)n * KCAT + kb + hi * 8);
            acc[0][t] = __builtin_amdgcn_wmma_f32_16x16x32_bf16(
                false, a0.v, false, b.v, (short)0, acc[0][t], false, false);
            acc[1][t] = __builtin_amdgcn_wmma_f32_16x16x32_bf16(
                false, a1.v, false, b.v, (short)0, acc[1][t], false, false);
        }
    }
    __syncthreads();   // sA dead; sH (aliased) writable

#pragma unroll
    for (int t = 0; t < 4; ++t) {
        const int c  = wave * 64 + t * 16 + r16;
        const float bb = b1[c];
#pragma unroll
        for (int m = 0; m < 2; ++m) {
#pragma unroll
            for (int v = 0; v < 8; ++v) {
                const int r = m * 16 + v + hi * 8;
                float x = acc[m][t][v] + bb;
                sH[r][c] = x > 0.f ? x : 0.f;
            }
        }
    }
    __syncthreads();

    {
        const int r = tid >> 2, sub = tid & 3;
        float sum = 0.f, sq = 0.f;
        for (int c = sub * 64; c < sub * 64 + 64; ++c) {
            float x = sH[r][c]; sum += x; sq += x * x;
        }
        sRed[r][sub][0] = sum; sRed[r][sub][1] = sq;
    }
    __syncthreads();
    if (tid < MROWS) {
        float sum = 0.f, sq = 0.f;
        for (int i = 0; i < 4; ++i) { sum += sRed[tid][i][0]; sq += sRed[tid][i][1]; }
        float mu  = sum * (1.f / DFEAT);
        float var = sq * (1.f / DFEAT) - mu * mu;
        sMu[tid] = mu;
        sRs[tid] = rsqrtf(var + 1e-5f);
    }
    __syncthreads();
    for (int i = tid; i < MROWS * DFEAT; i += 128) {
        int r = i >> 8, c = i & 255;
        float x = (sH[r][c] - sMu[r]) * sRs[r] * gamma[c] + beta[c];
        sHb[r][c] = f2bf(x);
    }
    __syncthreads();

    v8f acc2[2][4];
#pragma unroll
    for (int m = 0; m < 2; ++m)
#pragma unroll
        for (int t = 0; t < 4; ++t)
            acc2[m][t] = v8f{0.f,0.f,0.f,0.f,0.f,0.f,0.f,0.f};
    for (int kb = 0; kb < DFEAT; kb += 32) {
        BFrag a0 = load_a_frag(&sHb[r16][kb + hi * 8]);
        BFrag a1 = load_a_frag(&sHb[16 + r16][kb + hi * 8]);
#pragma unroll
        for (int t = 0; t < 4; ++t) {
            const int n = wave * 64 + t * 16 + r16;
            BFrag b = load_b_frag(W2t + (size_t)n * DFEAT + kb + hi * 8);
            acc2[0][t] = __builtin_amdgcn_wmma_f32_16x16x32_bf16(
                false, a0.v, false, b.v, (short)0, acc2[0][t], false, false);
            acc2[1][t] = __builtin_amdgcn_wmma_f32_16x16x32_bf16(
                false, a1.v, false, b.v, (short)0, acc2[1][t], false, false);
        }
    }

#pragma unroll
    for (int t = 0; t < 4; ++t) {
        const int c  = wave * 64 + t * 16 + r16;
        const float bb = b2[c];
#pragma unroll
        for (int m = 0; m < 2; ++m) {
#pragma unroll
            for (int v = 0; v < 8; ++v) {
                const int r  = m * 16 + v + hi * 8;
                const int nd = n0 + r;
                if (nd < NTOT)
                    nodes_out[(size_t)nd * DFEAT + c] = acc2[m][t][v] + bb;
            }
        }
    }
}

// ---------------------------------------------------------------------------
extern "C" void kernel_launch(void* const* d_in, const int* in_sizes, int n_in,
                              void* d_out, int out_size, void* d_ws, size_t ws_size,
                              hipStream_t stream) {
    (void)in_sizes; (void)n_in; (void)out_size; (void)ws_size;

    const float* nodes     = (const float*)d_in[0];
    const float* edges     = (const float*)d_in[1];
    const int*   senders   = (const int*)d_in[2];
    const int*   receivers = (const int*)d_in[3];
    const float* W1e = (const float*)d_in[4];
    const float* b1e = (const float*)d_in[5];
    const float* ge  = (const float*)d_in[6];
    const float* oe  = (const float*)d_in[7];
    const float* W2e = (const float*)d_in[8];
    const float* b2e = (const float*)d_in[9];
    const float* W1n = (const float*)d_in[10];
    const float* b1n = (const float*)d_in[11];
    const float* gn  = (const float*)d_in[12];
    const float* on  = (const float*)d_in[13];
    const float* W2n = (const float*)d_in[14];
    const float* b2n = (const float*)d_in[15];
    float* out = (float*)d_out;

    // Workspace layout: agg (fp32, N*D) then bf16 transposed weights.
    char*  ws  = (char*)d_ws;
    float* agg = (float*)ws;
    size_t off = (size_t)NTOT * DFEAT * sizeof(float);       // 108.7 MB
    unsigned short* w1e_t = (unsigned short*)(ws + off); off += (size_t)NSTEP * 256 * 512 * 2;
    unsigned short* w2e_t = (unsigned short*)(ws + off); off += (size_t)NSTEP * 256 * 256 * 2;
    unsigned short* w1n_t = (unsigned short*)(ws + off); off += (size_t)NSTEP * 256 * 512 * 2;
    unsigned short* w2n_t = (unsigned short*)(ws + off);

    convert_weights<<<dim3(512, 3, 4), 256, 0, stream>>>(
        W1e, W2e, W1n, W2n, w1e_t, w2e_t, w1n_t, w2n_t);

    const int n4 = NTOT * DFEAT / 4;
    for (int s = 0; s < NSTEP; ++s) {
        zero_f4<<<(n4 + 255) / 256, 256, 0, stream>>>((float4*)agg, n4);

        const float* nsrc = (s == 0) ? nodes : out;
        edge_mlp_scatter<<<NEDGE / MROWS, 128, 0, stream>>>(
            nsrc, edges, senders, receivers,
            w1e_t + (size_t)s * 256 * 512,
            W1e + (size_t)s * 515 * 256,          // rows 0..2 (raw edge features)
            b1e + (size_t)s * DFEAT,
            ge  + (size_t)s * DFEAT,
            oe  + (size_t)s * DFEAT,
            w2e_t + (size_t)s * 256 * 256,
            b2e + (size_t)s * DFEAT,
            agg);

        node_mlp<<<(NTOT + MROWS - 1) / MROWS, 128, 0, stream>>>(
            nsrc, agg,
            w1n_t + (size_t)s * 256 * 512,
            b1n + (size_t)s * DFEAT,
            gn  + (size_t)s * DFEAT,
            on  + (size_t)s * DFEAT,
            w2n_t + (size_t)s * 256 * 256,
            b2n + (size_t)s * DFEAT,
            out);
    }
}
